// RNNSpellChecker_7636451852651
// MI455X (gfx1250) — compile-verified, matrix operand order checked by reading, not run on hardware
//
#include <hip/hip_runtime.h>

#define B_  64
#define T_  512
#define E_  8
#define U_  512
#define G_  2048   // 4*U
#define V_  128
#define BT_ (B_*T_)

typedef __attribute__((ext_vector_type(16))) __bf16 bf16x16;
typedef __attribute__((ext_vector_type(8)))  float  f32x8;
typedef __attribute__((ext_vector_type(4)))  unsigned u32x4;
typedef __attribute__((ext_vector_type(8)))  unsigned u32x8;

union ABf { uint4 u[2]; bf16x16 v; };

// K index of the j-th packed half for a given lane (16-bit WMMA operand layout,
// ISA 7.12.2: lanes 0-15 hold K {0..7,16..23}, lanes 16-31 hold K {8..15,24..31})
__device__ __forceinline__ int kidx(int lane, int j) {
  return j + (j & 8) + ((lane & 16) >> 1);
}

__device__ __forceinline__ f32x8 zero8() {
  f32x8 z = {0.f,0.f,0.f,0.f,0.f,0.f,0.f,0.f};
  return z;
}
__device__ __forceinline__ float sigm(float x)  { return 1.0f / (1.0f + __expf(-x)); }
__device__ __forceinline__ float tanh_(float x) { return 1.0f - 2.0f / (__expf(2.0f * x) + 1.0f); }

__device__ __forceinline__ f32x8 wmma_bf16(const ABf& a, const ABf& b, f32x8 c) {
  return __builtin_amdgcn_wmma_f32_16x16x32_bf16(false, a.v, false, b.v, (short)0, c, false, false);
}

// Async global -> LDS copy, 16B per lane (CDNA5 GLOBAL_LOAD_ASYNC_TO_LDS_B128).
__device__ __forceinline__ void async_copy_b128(unsigned lds_off, const void* gsrc) {
  asm volatile("global_load_async_to_lds_b128 %0, %1, off"
               :: "v"(lds_off), "v"((unsigned long long)(size_t)gsrc)
               : "memory");
}
__device__ __forceinline__ void wait_async() {
  asm volatile("s_wait_asynccnt 0x0" ::: "memory");
}

// ---------------------------------------------------------------------------
__global__ void init_ws_k(unsigned* bar) {
  if (threadIdx.x < 24) bar[threadIdx.x] = 0u;
}

// ---------------------------------------------------------------------------
// Pack a row-major f32 [K,N] weight matrix into the bf16 WMMA B-operand layout:
// tile (kt,nt) -> 32 lanes x 16 halves, lane holds column nt*16+(lane&15),
// halves at K = kt*32 + kidx(lane,j).  Tile order: tile = kt*(N/16) + nt.
// ---------------------------------------------------------------------------
__global__ void pack_b(const float* __restrict__ src, __bf16* __restrict__ dst,
                       int K, int N) {
  int gid  = blockIdx.x * blockDim.x + threadIdx.x;
  int lane = gid & 31;
  int tile = gid >> 5;
  int ktiles = K >> 5, ntiles = N >> 4;
  if (tile >= ktiles * ntiles) return;
  int kt = tile / ntiles, nt = tile % ntiles;
  int n  = nt * 16 + (lane & 15);
  __attribute__((aligned(16))) __bf16 tmp[16];
#pragma unroll
  for (int j = 0; j < 16; ++j) {
    int k = kt * 32 + kidx(lane, j);
    tmp[j] = (__bf16)src[(size_t)k * N + n];
  }
  const uint4* t4 = (const uint4*)tmp;
  uint4* dp = (uint4*)(dst + ((size_t)tile * 32 + lane) * 16);
  dp[0] = t4[0];
  dp[1] = t4[1];
}

// ---------------------------------------------------------------------------
// Layer-0 input projection: xw0[bt,:] = emb[tok[bt]] @ W0 + b0  (K = E = 8).
// ---------------------------------------------------------------------------
__global__ __launch_bounds__(256) void embed_xw0(const int* __restrict__ tok,
                                                 const float* __restrict__ emb,
                                                 const float* __restrict__ W0,
                                                 const float* __restrict__ b0,
                                                 __bf16* __restrict__ xw) {
  int bt = blockIdx.x;
  int c0 = threadIdx.x << 3;
  int tk = tok[bt];
  float e[E_];
#pragma unroll
  for (int k = 0; k < E_; ++k) e[k] = emb[tk * E_ + k];
#pragma unroll
  for (int j = 0; j < 8; ++j) {
    int col = c0 + j;
    float acc = b0[col];
#pragma unroll
    for (int k = 0; k < E_; ++k) acc = fmaf(e[k], W0[k * G_ + col], acc);
    xw[(size_t)bt * G_ + col] = (__bf16)acc;
  }
}

// ---------------------------------------------------------------------------
// Parallel projection GEMM: xw = yin @ Wpk + bias  (M=BT=32768, N=2048, K=512)
// The 64KB of B tiles for this WG's 64 columns are staged once into LDS with
// async global->LDS loads and shared by all 8 waves.
// ---------------------------------------------------------------------------
__global__ __launch_bounds__(256) void proj_gemm(const __bf16* __restrict__ yin,
                                                 const __bf16* __restrict__ wpk,
                                                 const float* __restrict__ bias,
                                                 __bf16* __restrict__ xw) {
  extern __shared__ __align__(16) unsigned char lds_raw[];
  uint4* lds4 = (uint4*)lds_raw;
  unsigned ldsbase = (unsigned)(size_t)lds_raw;
  int tid = threadIdx.x, lane = tid & 31, wv = tid >> 5;
  int nb = blockIdx.x;                        // 0..31 (64 cols each)

  // Stage 64 B tiles: lds tile lt = kt*4 + i  <-  global tile kt*128 + nb*4 + i
  for (int idx = tid; idx < 4096; idx += 256) {
    int lt = idx >> 6, off = idx & 63;
    int kt = lt >> 2, i = lt & 3;
    int gt = kt * 128 + nb * 4 + i;
    async_copy_b128(ldsbase + idx * 16, (const uint4*)wpk + (size_t)gt * 64 + off);
  }
  wait_async();
  __syncthreads();

  int m0 = blockIdx.y * 128 + wv * 16;
  int arow = m0 + (lane & 15);
  int asub = (lane >> 4) << 3;
  f32x8 acc[4];
#pragma unroll
  for (int i = 0; i < 4; ++i) acc[i] = zero8();

#pragma unroll
  for (int kt = 0; kt < 16; ++kt) {
    ABf a;
    const __bf16* ap = yin + (size_t)arow * U_ + kt * 32 + asub;
    a.u[0] = *(const uint4*)ap;
    a.u[1] = *(const uint4*)(ap + 16);
#pragma unroll
    for (int i = 0; i < 4; ++i) {
      ABf b;
      int tb = (kt * 4 + i) * 64 + lane * 2;
      b.u[0] = lds4[tb];
      b.u[1] = lds4[tb + 1];
      acc[i] = wmma_bf16(a, b, acc[i]);
    }
  }
#pragma unroll
  for (int i = 0; i < 4; ++i) {
    int col = (nb * 4 + i) * 16 + (lane & 15);
    float bv = bias[col];
#pragma unroll
    for (int r = 0; r < 8; ++r) {
      int row = m0 + r + ((lane >> 4) << 3);
      xw[(size_t)row * G_ + col] = (__bf16)(acc[i][r] + bv);
    }
  }
}

// ---------------------------------------------------------------------------
// Persistent recurrent LSTM layer.  16 WGs (one per WGP); WG w owns h-columns
// [32w,32w+32).  Its 128KB U slice is DMA'd into LDS by the Tensor Data Mover
// as one 3-D tile (x = 2KB run of 2 column-tiles, y = 4 gates @ 32KB stride,
// z = 16 k-tiles @ 128KB stride) and stays stationary for all 512 steps.
// c stays in VGPRs; h ping-pongs through L2 via y[b][t][col].
// ---------------------------------------------------------------------------
__global__ __launch_bounds__(256) void lstm_layer(const __bf16* __restrict__ xw,
                                                  const __bf16* __restrict__ upk,
                                                  __bf16* __restrict__ y,
                                                  unsigned* __restrict__ bar) {
  extern __shared__ __align__(16) unsigned char lds_raw[];
  uint4* lds4 = (uint4*)lds_raw;
  int tid = threadIdx.x, lane = tid & 31, wv = tid >> 5, w = blockIdx.x;

  if (wv == 0) {
    // Build the Tensor DMA Descriptor (D#) in SGPR groups.
    unsigned long long ga = (unsigned long long)(size_t)upk + (size_t)w * 2048;
    unsigned lds_addr = (unsigned)(size_t)lds_raw;
    u32x4 g0;
    g0.x = 1u;                                      // count=1, user descriptor
    g0.y = lds_addr;                                // lds_addr[31:0]
    g0.z = (unsigned)ga;                            // global_addr[31:0]
    g0.w = (unsigned)((ga >> 32) & 0x01FFFFFFull) | 0x80000000u;  // addr[56:32] | type=2
    u32x8 g1;
    g1.s0 = 0x00030000u;                            // data_size=3 (8B), no mask/pad
    g1.s1 = 0x00000000u;                            // abar=0, tensor_dim0[15:0]=0
    g1.s2 = 0x00004000u;                            // tensor_dim0=2^30, tensor_dim1 lo
    g1.s3 = 0x01004000u;                            // tensor_dim1=2^30, tile_dim0=256
    g1.s4 = 0x00100004u;                            // tile_dim1=4, tile_dim2=16
    g1.s5 = 4096u;                                  // dim0_stride = 32KB/8B
    g1.s6 = 0x40000000u;                            // dim1_stride = 128KB/8B (lo16<<16)
    g1.s7 = 0u;
    u32x4 g2;
    g2.x = 0x40000000u;                             // tensor_dim2 = 2^30
    g2.y = 0u; g2.z = 0u; g2.w = 0u;                // dim3/stride2/tile_dim3 unused
    u32x4 g3 = {0u, 0u, 0u, 0u};
    asm volatile("tensor_load_to_lds %0, %1, %2, %3"
                 :: "s"(g0), "s"(g1), "s"(g2), "s"(g3) : "memory");
    __builtin_amdgcn_s_wait_tensorcnt(0);
  }
  __syncthreads();

  int mt = wv >> 1, ntl = wv & 1;                  // wave = (m-tile, n-tile)
  int m0 = mt * 16;
  int hcol  = w * 32 + ntl * 16 + (lane & 15);     // this lane's h column
  int arow  = m0 + (lane & 15);                    // A-operand batch row
  int asub  = (lane >> 4) << 3;
  int rbase = m0 + ((lane >> 4) << 3);             // C-operand base batch row

  f32x8 c = zero8();                               // cell state lives in VGPRs
  unsigned tgt = 0;

  for (int t = 0; t < T_; ++t) {
    f32x8 zi = zero8(), zf = zero8(), zg = zero8(), zo = zero8();
    if (t > 0) {
      const __bf16* hp = y + ((size_t)arow * T_ + (t - 1)) * U_;
#pragma unroll
      for (int kt = 0; kt < 16; ++kt) {
        ABf a;
        const __bf16* ap = hp + kt * 32 + asub;
        a.u[0] = *(const uint4*)ap;
        a.u[1] = *(const uint4*)(ap + 16);
        // TDM linear order: uint4 index = (kt*4 + gate)*128 + ntl*64 + lane*2
        int tb = kt * 512 + ntl * 64 + lane * 2;
        ABf bI, bF, bG, bO;
        bI.u[0] = lds4[tb];       bI.u[1] = lds4[tb + 1];
        bF.u[0] = lds4[tb + 128]; bF.u[1] = lds4[tb + 129];
        bG.u[0] = lds4[tb + 256]; bG.u[1] = lds4[tb + 257];
        bO.u[0] = lds4[tb + 384]; bO.u[1] = lds4[tb + 385];
        zi = wmma_bf16(a, bI, zi);
        zf = wmma_bf16(a, bF, zf);
        zg = wmma_bf16(a, bG, zg);
        zo = wmma_bf16(a, bO, zo);
      }
    }
    // Elementwise LSTM cell: z = xw + h@U (bias already folded into xw).
#pragma unroll
    for (int r = 0; r < 8; ++r) {
      size_t xo_ = ((size_t)(rbase + r) * T_ + t) * (size_t)G_;
      float xi  = (float)xw[xo_ + hcol];
      float xf  = (float)xw[xo_ + hcol + 512];
      float xg  = (float)xw[xo_ + hcol + 1024];
      float xo2 = (float)xw[xo_ + hcol + 1536];
      float iv = sigm(zi[r] + xi);
      float fv = sigm(zf[r] + xf);
      float gv = tanh_(zg[r] + xg);
      float ov = sigm(zo[r] + xo2);
      float cv = fv * c[r] + iv * gv;
      c[r] = cv;
      float hv = ov * tanh_(cv);
      y[((size_t)(rbase + r) * T_ + t) * U_ + hcol] = (__bf16)hv;
    }
    if (t + 1 < T_)
      __builtin_prefetch(xw + ((size_t)rbase * T_ + (t + 1)) * (size_t)G_ + hcol, 0, 1);

    // Device-scope barrier across the 16 co-resident WGs.
    __threadfence();
    __syncthreads();
    tgt += 16;
    if (tid == 0) {
      __hip_atomic_fetch_add(bar, 1u, __ATOMIC_ACQ_REL, __HIP_MEMORY_SCOPE_AGENT);
      while (__hip_atomic_load(bar, __ATOMIC_ACQUIRE, __HIP_MEMORY_SCOPE_AGENT) < tgt)
        __builtin_amdgcn_s_sleep(1);
    }
    __syncthreads();
    __threadfence();
  }
}

// ---------------------------------------------------------------------------
// Final dense + softmax: logits = y2 @ Wd + bd, softmax over V=128.
// Packed Wd (128KB) staged via async global->LDS loads.
// ---------------------------------------------------------------------------
__global__ __launch_bounds__(256) void dense_softmax(const __bf16* __restrict__ y2,
                                                     const __bf16* __restrict__ wdpk,
                                                     const float* __restrict__ bd,
                                                     float* __restrict__ out) {
  extern __shared__ __align__(16) unsigned char lds_raw[];
  uint4* lds4 = (uint4*)lds_raw;
  unsigned ldsbase = (unsigned)(size_t)lds_raw;
  int tid = threadIdx.x, lane = tid & 31, wv = tid >> 5;
  for (int i = tid; i < 8192; i += 256)
    async_copy_b128(ldsbase + i * 16, (const uint4*)wdpk + i);
  wait_async();
  __syncthreads();

  int m0 = blockIdx.x * 128 + wv * 16;
  int arow = m0 + (lane & 15);
  int asub = (lane >> 4) << 3;
  f32x8 acc[8];
#pragma unroll
  for (int i = 0; i < 8; ++i) acc[i] = zero8();

#pragma unroll
  for (int kt = 0; kt < 16; ++kt) {
    ABf a;
    const __bf16* ap = y2 + (size_t)arow * U_ + kt * 32 + asub;
    a.u[0] = *(const uint4*)ap;
    a.u[1] = *(const uint4*)(ap + 16);
#pragma unroll
    for (int nt = 0; nt < 8; ++nt) {
      ABf b;
      int tb = (kt * 8 + nt) * 64 + lane * 2;
      b.u[0] = lds4[tb];
      b.u[1] = lds4[tb + 1];
      acc[nt] = wmma_bf16(a, b, acc[nt]);
    }
  }
#pragma unroll
  for (int nt = 0; nt < 8; ++nt) {
    float bv = bd[nt * 16 + (lane & 15)];
#pragma unroll
    for (int r = 0; r < 8; ++r) acc[nt][r] += bv;
  }

  int rbase = m0 + ((lane >> 4) << 3);
#pragma unroll
  for (int r = 0; r < 8; ++r) {
    float mx = acc[0][r];
#pragma unroll
    for (int nt = 1; nt < 8; ++nt) mx = fmaxf(mx, acc[nt][r]);
#pragma unroll
    for (int m = 8; m >= 1; m >>= 1) mx = fmaxf(mx, __shfl_xor(mx, m, 32));
    float e[8];
    float s = 0.f;
#pragma unroll
    for (int nt = 0; nt < 8; ++nt) { e[nt] = __expf(acc[nt][r] - mx); s += e[nt]; }
#pragma unroll
    for (int m = 8; m >= 1; m >>= 1) s += __shfl_xor(s, m, 32);
    float inv = 1.0f / s;
    size_t ro = (size_t)(rbase + r) * V_;
#pragma unroll
    for (int nt = 0; nt < 8; ++nt)
      out[ro + nt * 16 + (lane & 15)] = e[nt] * inv;
  }
}

// ---------------------------------------------------------------------------
extern "C" void kernel_launch(void* const* d_in, const int* in_sizes, int n_in,
                              void* d_out, int out_size, void* d_ws, size_t ws_size,
                              hipStream_t stream) {
  (void)in_sizes; (void)n_in; (void)out_size; (void)ws_size;
  const int*   tok = (const int*)d_in[0];
  const float* emb = (const float*)d_in[1];
  const float* W0  = (const float*)d_in[2];
  const float* Ur0 = (const float*)d_in[3];
  const float* b0  = (const float*)d_in[4];
  const float* W1  = (const float*)d_in[5];
  const float* Ur1 = (const float*)d_in[6];
  const float* b1  = (const float*)d_in[7];
  const float* W2  = (const float*)d_in[8];
  const float* Ur2 = (const float*)d_in[9];
  const float* b2  = (const float*)d_in[10];
  const float* Wd  = (const float*)d_in[11];
  const float* bd  = (const float*)d_in[12];
  float* out = (float*)d_out;

  char* ws = (char*)d_ws;
  size_t off = 0;
  __bf16* XW  = (__bf16*)(ws + off); off += (size_t)BT_ * G_ * 2;   // 134 MB
  __bf16* Y0  = (__bf16*)(ws + off); off += (size_t)BT_ * U_ * 2;   //  33 MB
  __bf16* Y1  = (__bf16*)(ws + off); off += (size_t)BT_ * U_ * 2;   //  33 MB
  __bf16* U0p = (__bf16*)(ws + off); off += (size_t)U_ * G_ * 2;
  __bf16* U1p = (__bf16*)(ws + off); off += (size_t)U_ * G_ * 2;
  __bf16* U2p = (__bf16*)(ws + off); off += (size_t)U_ * G_ * 2;
  __bf16* W1p = (__bf16*)(ws + off); off += (size_t)U_ * G_ * 2;
  __bf16* W2p = (__bf16*)(ws + off); off += (size_t)U_ * G_ * 2;
  __bf16* Wdp = (__bf16*)(ws + off); off += (size_t)U_ * V_ * 2;
  unsigned* bar = (unsigned*)(ws + off); off += 256;

  hipFuncSetAttribute(reinterpret_cast<const void*>(lstm_layer),
                      hipFuncAttributeMaxDynamicSharedMemorySize, 131072);
  hipFuncSetAttribute(reinterpret_cast<const void*>(dense_softmax),
                      hipFuncAttributeMaxDynamicSharedMemorySize, 131072);
  hipFuncSetAttribute(reinterpret_cast<const void*>(proj_gemm),
                      hipFuncAttributeMaxDynamicSharedMemorySize, 65536);

  init_ws_k<<<1, 32, 0, stream>>>(bar);

  // Pre-swizzle weights into WMMA B-operand layout (bf16).
  pack_b<<<256, 256, 0, stream>>>(Ur0, U0p, U_, G_);
  pack_b<<<256, 256, 0, stream>>>(Ur1, U1p, U_, G_);
  pack_b<<<256, 256, 0, stream>>>(Ur2, U2p, U_, G_);
  pack_b<<<256, 256, 0, stream>>>(W1,  W1p, U_, G_);
  pack_b<<<256, 256, 0, stream>>>(W2,  W2p, U_, G_);
  pack_b<<<16,  256, 0, stream>>>(Wd,  Wdp, U_, V_);

  // Layer 0 input projection (K=8) fused with embedding gather.
  embed_xw0<<<BT_, 256, 0, stream>>>(tok, emb, W0, b0, XW);

  // Layer 0 recurrence -> Y0
  lstm_layer<<<16, 256, 131072, stream>>>(XW, U0p, Y0, bar + 0);
  // Layer 1: projection then recurrence -> Y1
  proj_gemm<<<dim3(32, 256), 256, 65536, stream>>>(Y0, W1p, b1, XW);
  lstm_layer<<<16, 256, 131072, stream>>>(XW, U1p, Y1, bar + 8);
  // Layer 2: projection then recurrence -> Y0 (reused)
  proj_gemm<<<dim3(32, 256), 256, 65536, stream>>>(Y1, W2p, b2, XW);
  lstm_layer<<<16, 256, 131072, stream>>>(XW, U2p, Y0, bar + 16);
  // Dense + softmax
  dense_softmax<<<256, 256, 131072, stream>>>(Y0, Wdp, bd, out);
}
